// SparseLinearAttention_4440996184229
// MI455X (gfx1250) — compile-verified, matrix operand order checked by reading, not run on hardware
//
#include <hip/hip_runtime.h>
#include <hip/hip_bf16.h>

typedef __attribute__((ext_vector_type(16))) __bf16 v16bf;
typedef __attribute__((ext_vector_type(8)))  __bf16 v8bf;
typedef __attribute__((ext_vector_type(8)))  float  v8f;

static constexpr int H_    = 16;
static constexpr int D_    = 64;
static constexpr int L_    = 4096;
static constexpr int MBLK  = 64;   // 64-row blocks along L
static constexpr int TK    = 8;    // top-k key blocks
static constexpr int RS    = 72;   // LDS row stride (bf16 elems); 144B keeps 16B alignment

static __device__ __forceinline__ v8f wmma_bf16(v16bf a, v16bf b, v8f c) {
  return __builtin_amdgcn_wmma_f32_16x16x32_bf16(false, a, false, b, (short)0, c, false, false);
}

// Row-style fragment: lane = row (p = &lds[row*RS + C0]); K elements contiguous:
// K = 8*half + {0..7} and 16 + 8*half + {0..7}  -> two 16B LDS loads.
static __device__ __forceinline__ v16bf frag_row(const __bf16* p, int hf) {
  v8bf x = *(const v8bf*)(p + 8 * hf);
  v8bf y = *(const v8bf*)(p + 16 + 8 * hf);
  v16bf r;
#pragma unroll
  for (int i = 0; i < 8; ++i) { r[i] = x[i]; r[i + 8] = y[i]; }
  return r;
}

// 32 contiguous f32 from global -> bf16 row in LDS
static __device__ __forceinline__ void stage32(const float* g, __bf16* l) {
#pragma unroll
  for (int i = 0; i < 8; ++i) {
    float4 f = ((const float4*)g)[i];
    l[4*i+0] = (__bf16)f.x; l[4*i+1] = (__bf16)f.y;
    l[4*i+2] = (__bf16)f.z; l[4*i+3] = (__bf16)f.w;
  }
}

// 32 contiguous f32 from global -> bf16 COLUMN in LDS (transposed staging):
// l points at &T[c0*RS + row]; element j goes to T[(c0+j)*RS + row]
static __device__ __forceinline__ void stage32t(const float* g, __bf16* l) {
#pragma unroll
  for (int i = 0; i < 8; ++i) {
    float4 f = ((const float4*)g)[i];
    l[(4*i+0)*RS] = (__bf16)f.x; l[(4*i+1)*RS] = (__bf16)f.y;
    l[(4*i+2)*RS] = (__bf16)f.z; l[(4*i+3)*RS] = (__bf16)f.w;
  }
}

// ---------------- Kernel 1: block mean descriptors ----------------
__global__ void kmeans_kernel(const float* __restrict__ q, const float* __restrict__ k,
                              float* __restrict__ qm, float* __restrict__ km) {
  int id = blockIdx.x;
  bool isK = id >= H_ * MBLK;
  int hm = isK ? id - H_ * MBLK : id;
  int h = hm / MBLK, m = hm % MBLK;
  const float* src = isK ? k : q;
  float* dst = isK ? km : qm;
  int d = threadIdx.x;
  float s = 0.f;
  for (int r = 0; r < 64; ++r)
    s += src[(size_t)((m * 64 + r) * H_ + h) * D_ + d];
  dst[(h * MBLK + m) * D_ + d] = s * (1.f / 64.f);
}

// ---------------- Kernel 2: block scores (fp32 exact) + top-8 ----------------
__global__ void topk_kernel(const float* __restrict__ qm, const float* __restrict__ km,
                            int* __restrict__ lut) {
  __shared__ float qs[64 * 64];
  __shared__ float ks[64 * 64];
  int h = blockIdx.x, t = threadIdx.x;
  for (int d = 0; d < 64; ++d) {
    qs[t * 64 + d] = qm[(h * MBLK + t) * D_ + d];
    ks[t * 64 + d] = km[(h * MBLK + t) * D_ + d];
  }
  __syncthreads();
  float sc[64];
  for (int n = 0; n < 64; ++n) {
    float s = 0.f;
    for (int d = 0; d < 64; ++d) s += qs[t * 64 + d] * ks[n * 64 + d];
    sc[n] = s;
  }
  for (int j = 0; j < TK; ++j) {           // strict '>' => lowest index on ties (jax top_k)
    int best = 0; float bv = sc[0];
    for (int n = 1; n < 64; ++n) if (sc[n] > bv) { bv = sc[n]; best = n; }
    lut[(h * MBLK + t) * TK + j] = best;
    sc[best] = -3.0e38f;
  }
}

// ---------------- Kernel 3: block-sparse flash attention (bf16 WMMA) ----------------
__global__ __launch_bounds__(128) void sparse_attn_kernel(
    const float* __restrict__ q, const float* __restrict__ k, const float* __restrict__ v,
    const int* __restrict__ lut, float* __restrict__ out) {
  __shared__ __align__(16) __bf16 Qs[64 * RS];   // [q-row][d]
  __shared__ __align__(16) __bf16 Ks[64 * RS];   // [key][d]   (row-style for S B-frag)
  __shared__ __align__(16) __bf16 Vt[64 * RS];   // [d][key]   (transposed: row-style for PV B-frag)
  __shared__ __align__(16) __bf16 Ps[64 * RS];   // [q-row][key]
  int h = blockIdx.x >> 6, m = blockIdx.x & 63;
  int tid = threadIdx.x, w = tid >> 5, lane = tid & 31;
  int n = lane & 15, hf = lane >> 4;
  int sr = tid >> 1, sc0 = (tid & 1) * 32;

  stage32(q + (size_t)((m * 64 + sr) * H_ + h) * D_ + sc0, &Qs[sr * RS + sc0]);

  float rm[8], rsum[8];
  v8f acc[4] = {};
#pragma unroll
  for (int r = 0; r < 8; ++r) { rm[r] = -1e30f; rsum[r] = 0.f; }

  for (int t = 0; t < TK; ++t) {
    __syncthreads();                       // prior tile's compute done (also covers Q staging)
    int nb = lut[(h * MBLK + m) * TK + t];
    stage32 (k + (size_t)((nb * 64 + sr) * H_ + h) * D_ + sc0, &Ks[sr * RS + sc0]);
    stage32t(v + (size_t)((nb * 64 + sr) * H_ + h) * D_ + sc0, &Vt[sc0 * RS + sr]);
    if (t + 1 < TK) {                      // prefetch next gathered block (global_prefetch_b8)
      int nb2 = lut[(h * MBLK + m) * TK + t + 1];
      __builtin_prefetch(k + (size_t)((nb2 * 64 + sr) * H_ + h) * D_ + sc0, 0, 1);
      __builtin_prefetch(v + (size_t)((nb2 * 64 + sr) * H_ + h) * D_ + sc0, 0, 1);
    }
    __syncthreads();

    // S = (Q * K^T) * D^-0.5   (16 rows per wave x 64 keys)
    v16bf qa0 = frag_row(&Qs[(16 * w + n) * RS + 0],  hf);
    v16bf qa1 = frag_row(&Qs[(16 * w + n) * RS + 32], hf);
    v8f s[4];
#pragma unroll
    for (int nt = 0; nt < 4; ++nt) {
      v8f sv = {};
      sv = wmma_bf16(qa0, frag_row(&Ks[(16 * nt + n) * RS + 0],  hf), sv);
      sv = wmma_bf16(qa1, frag_row(&Ks[(16 * nt + n) * RS + 32], hf), sv);
      s[nt] = sv * 0.125f;
    }
    // online softmax (fp32 stats; row = r + 8*hf within the wave's 16-row stripe)
#pragma unroll
    for (int r = 0; r < 8; ++r) {
      float tmax = fmaxf(fmaxf(s[0][r], s[1][r]), fmaxf(s[2][r], s[3][r]));
      tmax = fmaxf(tmax, __shfl_xor(tmax, 1, 32));
      tmax = fmaxf(tmax, __shfl_xor(tmax, 2, 32));
      tmax = fmaxf(tmax, __shfl_xor(tmax, 4, 32));
      tmax = fmaxf(tmax, __shfl_xor(tmax, 8, 32));
      float mnew = fmaxf(rm[r], tmax);
      float f = __expf(rm[r] - mnew);
      rm[r] = mnew;
      float psum = 0.f;
#pragma unroll
      for (int nt = 0; nt < 4; ++nt) {
        float p = __expf(s[nt][r] - mnew);
        Ps[(16 * w + r + 8 * hf) * RS + 16 * nt + n] = (__bf16)p;   // C-layout -> LDS
        psum += p;
        acc[nt][r] *= f;
      }
      psum += __shfl_xor(psum, 1, 32);
      psum += __shfl_xor(psum, 2, 32);
      psum += __shfl_xor(psum, 4, 32);
      psum += __shfl_xor(psum, 8, 32);
      rsum[r] = rsum[r] * f + psum;
    }
    // O += P @ V (P reloaded in A-layout; V transposed in LDS -> row-style B-frags)
#pragma unroll
    for (int ks = 0; ks < 2; ++ks) {
      v16bf pa = frag_row(&Ps[(16 * w + n) * RS + 32 * ks], hf);
#pragma unroll
      for (int nt = 0; nt < 4; ++nt)
        acc[nt] = wmma_bf16(pa, frag_row(&Vt[(16 * nt + n) * RS + 32 * ks], hf), acc[nt]);
    }
  }
#pragma unroll
  for (int nt = 0; nt < 4; ++nt)
#pragma unroll
    for (int r = 0; r < 8; ++r) {
      int row = 16 * w + r + 8 * hf;
      int l = m * 64 + row;
      out[(size_t)(l * H_ + h) * D_ + 16 * nt + n] = acc[nt][r] / rsum[r];
    }
}

// ---------------- Kernel 4: kvt = (k_fm^T @ v)^T, ksum (per head) ----------------
__global__ __launch_bounds__(128) void kv_kernel(
    const float* __restrict__ k, const float* __restrict__ v,
    float* __restrict__ kvt, float* __restrict__ ksumws) {
  __shared__ __align__(16) __bf16 kft[32 * RS];  // [d][l]  (transposed k_fm chunk)
  __shared__ __align__(16) __bf16 vft[64 * RS];  // [e][l]  (transposed v chunk)
  int h = blockIdx.x;
  int tid = threadIdx.x, w = tid >> 5, lane = tid & 31;
  int n = lane & 15, hf = lane >> 4;
  v8f acc[4] = {};
  float kacc = 0.f;
  for (int c = 0; c < L_ / 32; ++c) {
    __syncthreads();
    if (tid < 32) {                         // k row softmax over D (fp32), store transposed
      int l = c * 32 + tid;
      const float* kr = k + (size_t)(l * H_ + h) * D_;
      float buf[64]; float mx = -1e30f;
#pragma unroll
      for (int d = 0; d < 64; ++d) { buf[d] = kr[d]; mx = fmaxf(mx, buf[d]); }
      float se = 0.f;
#pragma unroll
      for (int d = 0; d < 64; ++d) { buf[d] = __expf(buf[d] - mx); se += buf[d]; }
      float inv = 1.f / se;
#pragma unroll
      for (int d = 0; d < 64; ++d) kft[d * RS + tid] = (__bf16)(buf[d] * inv);
    } else if (tid < 96) {                  // stage v chunk transposed
      int idx = tid - 32, vr = idx >> 1, c0 = (idx & 1) * 32;
      int l = c * 32 + vr;
      stage32t(v + (size_t)(l * H_ + h) * D_ + c0, &vft[c0 * RS + vr]);
    }
    __syncthreads();
    if (tid < 64) {                         // ksum[d] partial (contiguous LDS row read)
#pragma unroll
      for (int r = 0; r < 32; ++r) kacc += (float)kft[tid * RS + r];
    }
    // A = k_fm^T (M = d-rows 16w..16w+15, K = 32 l's), B = v chunk; both row-style now
    v16bf a = frag_row(&kft[(16 * w + n) * RS], hf);
#pragma unroll
    for (int nt = 0; nt < 4; ++nt)
      acc[nt] = wmma_bf16(a, frag_row(&vft[(16 * nt + n) * RS], hf), acc[nt]);
  }
  // write kv TRANSPOSED: kvt[e][d]  (so kernel 5's B-frags are row-style)
#pragma unroll
  for (int nt = 0; nt < 4; ++nt)
#pragma unroll
    for (int r = 0; r < 8; ++r)
      kvt[(size_t)h * 4096 + (16 * nt + n) * 64 + (16 * w + r + 8 * hf)] = acc[nt][r];
  if (tid < 64) ksumws[h * 64 + tid] = kacc;
}

// ---------------- Kernel 5: o_l = ((q_fm @ kv)/den) @ Wp^T + b, added to out ----------------
__global__ __launch_bounds__(128) void linear_out_kernel(
    const float* __restrict__ q, const float* __restrict__ kvt,
    const float* __restrict__ ksumws, const float* __restrict__ wproj,
    const float* __restrict__ bproj, float* __restrict__ out) {
  __shared__ __align__(16) __bf16 qf[64 * RS];   // [row][d]
  __shared__ __align__(16) __bf16 kvb[64 * RS];  // [e][d]  (kv transposed -> row-style B)
  __shared__ __align__(16) __bf16 wps[64 * RS];  // [e][d]  (Wp as stored -> row-style B)
  __shared__ __align__(16) __bf16 ols[64 * RS];  // [row][d]
  __shared__ float sks[64];
  __shared__ float dens[64];
  int h = blockIdx.x >> 6, m = blockIdx.x & 63;
  int tid = threadIdx.x, w = tid >> 5, lane = tid & 31;
  int n = lane & 15, hf = lane >> 4;
  int sr = tid >> 1, c0 = (tid & 1) * 32;

  stage32(kvt + (size_t)h * 4096 + sr * 64 + c0, &kvb[sr * RS + c0]);
  stage32(wproj + sr * 64 + c0, &wps[sr * RS + c0]);
  if (tid >= 64) sks[tid - 64] = ksumws[h * 64 + tid - 64];
  __syncthreads();

  if (tid < 64) {                           // q_fm softmax + den (fp32)
    int l = m * 64 + tid;
    const float* qr = q + (size_t)(l * H_ + h) * D_;
    float buf[64]; float mx = -1e30f;
#pragma unroll
    for (int d = 0; d < 64; ++d) { buf[d] = qr[d]; mx = fmaxf(mx, buf[d]); }
    float se = 0.f;
#pragma unroll
    for (int d = 0; d < 64; ++d) { buf[d] = __expf(buf[d] - mx); se += buf[d]; }
    float inv = 1.f / se, den = 0.f;
#pragma unroll
    for (int d = 0; d < 64; ++d) {
      float p = buf[d] * inv;
      qf[tid * RS + d] = (__bf16)p;
      den += p * sks[d];
    }
    dens[tid] = den + 1e-6f;
  }
  __syncthreads();

  // GEMM1: num = q_fm @ kv   (B[kk=d][n=e] = kvt[e][d] -> row-style over kvb)
  v8f acc[4] = {};
#pragma unroll
  for (int ks = 0; ks < 2; ++ks) {
    v16bf a = frag_row(&qf[(16 * w + n) * RS + 32 * ks], hf);
#pragma unroll
    for (int nt = 0; nt < 4; ++nt)
      acc[nt] = wmma_bf16(a, frag_row(&kvb[(16 * nt + n) * RS + 32 * ks], hf), acc[nt]);
  }
#pragma unroll
  for (int nt = 0; nt < 4; ++nt)
#pragma unroll
    for (int r = 0; r < 8; ++r) {
      int row = 16 * w + r + 8 * hf;
      ols[row * RS + 16 * nt + n] = (__bf16)(acc[nt][r] / dens[row]);
    }

  // GEMM2: o_l @ Wp^T  (B[kk=d][n=e] = Wp[e][d] -> row-style over wps)
  v8f acc2[4] = {};
#pragma unroll
  for (int ks = 0; ks < 2; ++ks) {
    v16bf a = frag_row(&ols[(16 * w + n) * RS + 32 * ks], hf);
#pragma unroll
    for (int nt = 0; nt < 4; ++nt)
      acc2[nt] = wmma_bf16(a, frag_row(&wps[(16 * nt + n) * RS + 32 * ks], hf), acc2[nt]);
  }
#pragma unroll
  for (int nt = 0; nt < 4; ++nt)
#pragma unroll
    for (int r = 0; r < 8; ++r) {
      int row = 16 * w + r + 8 * hf;
      int l = m * 64 + row;
      size_t o = (size_t)(l * H_ + h) * D_ + 16 * nt + n;
      out[o] += acc2[nt][r] + bproj[16 * nt + n];
    }
}

extern "C" void kernel_launch(void* const* d_in, const int* in_sizes, int n_in,
                              void* d_out, int out_size, void* d_ws, size_t ws_size,
                              hipStream_t stream) {
  (void)in_sizes; (void)n_in; (void)out_size; (void)ws_size;
  const float* q  = (const float*)d_in[0];
  const float* k  = (const float*)d_in[1];
  const float* v  = (const float*)d_in[2];
  const float* wp = (const float*)d_in[3];
  const float* bp = (const float*)d_in[4];
  float* out = (float*)d_out;

  float* ws   = (float*)d_ws;
  float* qm   = ws;                 // 65536 f
  float* km   = ws + 65536;         // 65536 f
  float* kvt  = ws + 131072;        // 65536 f  (kv transposed: [h][e][d])
  float* ksum = ws + 196608;        // 64*16 f
  int*   lut  = (int*)(ws + 196672);// 8192 i32

  hipLaunchKernelGGL(kmeans_kernel,     dim3(2 * H_ * MBLK), dim3(64),  0, stream, q, k, qm, km);
  hipLaunchKernelGGL(topk_kernel,       dim3(H_),            dim3(64),  0, stream, qm, km, lut);
  hipLaunchKernelGGL(sparse_attn_kernel,dim3(H_ * MBLK),     dim3(128), 0, stream, q, k, v, lut, out);
  hipLaunchKernelGGL(kv_kernel,         dim3(H_),            dim3(128), 0, stream, k, v, kvt, ksum);
  hipLaunchKernelGGL(linear_out_kernel, dim3(H_ * MBLK),     dim3(128), 0, stream, q, kvt, ksum, wp, bp, out);
}
